// Transformer_88734024336131
// MI455X (gfx1250) — compile-verified, matrix operand order checked by reading, not run on hardware
//
#include <hip/hip_runtime.h>
#include <hip/hip_bf16.h>

// ---------------------------------------------------------------------------
// Types for WMMA / vector loads
// ---------------------------------------------------------------------------
typedef __attribute__((ext_vector_type(16))) _Float16 v16h;
typedef __attribute__((ext_vector_type(8)))  _Float16 v8h;
typedef __attribute__((ext_vector_type(8)))  float    v8f;
typedef __attribute__((ext_vector_type(4)))  float    v4f;

#define BM  128
#define BN  64
#define BKT 64
#define LS  72   // halfs per LDS row: 64 data + 8 pad (144B: 16B-aligned, bank-conflict free)

// ---------------------------------------------------------------------------
// Generic strided batched GEMM, fp32 in/out, f16 WMMA compute.
//   C[b,h] = alpha * A[b,h] @ B[b,h] (+bias) (+C if accum) (relu optional)
// A(m,k) at A + b*sAb + h*sAh + m*sAm + k*sAk   (same scheme for B(n,k), C(m,n))
// flags: 1 = relu, 2 = accumulate into C, 4 = add bias[col]
// MODE: 0 = guarded scalar staging (edge shapes: N=21 / K=21)
//       1 = fast path, A and B K-contiguous (sAk==sBk==1), M%128==0, K%64==0, N%64==0
//       2 = fast path, A K-contiguous, B N-contiguous (sBn==1)
// ---------------------------------------------------------------------------
template <int MODE>
__launch_bounds__(256)
__global__ void gemm_f16_wmma(
    const float* __restrict__ A, const float* __restrict__ B,
    const float* __restrict__ bias, float* __restrict__ C,
    int M, int N, int K,
    long sAb, long sAh, long sAm, long sAk,
    long sBb, long sBh, long sBn, long sBk,
    long sCb, long sCh, long sCm, long sCn,
    int H, float alpha, int flags)
{
    __shared__ __align__(16) _Float16 Asm[2][BM * LS];
    __shared__ __align__(16) _Float16 Bsm[2][BN * LS];

    const int bz = blockIdx.z;
    const int b  = bz / H;
    const int h  = bz % H;
    const float* Ab = A + (long)b * sAb + (long)h * sAh;
    const float* Bb = B + (long)b * sBb + (long)h * sBh;
    float*       Cb = C + (long)b * sCb + (long)h * sCh;

    const int m0 = blockIdx.y * BM;
    const int n0 = blockIdx.x * BN;
    const int t    = threadIdx.x;
    const int wave = t >> 5;
    const int lane = t & 31;
    const int wm = (wave & 3) * 32;   // 4 waves along M
    const int wn = (wave >> 2) * 32;  // 2 waves along N
    const int lrow = lane & 15;
    const int lgrp = (lane >> 4) * 8; // K sub-offset: lanes 0-15 -> 0, lanes 16-31 -> 8

    v8f acc[2][2] = {};

    // ---- compute 8 WMMAs on LDS buffer `buf` (two K=32 sub-steps) ----
    auto compute = [&](int buf) {
        #pragma unroll
        for (int s = 0; s < 2; ++s) {
            const int ko = s * 32;
            union Frag { v16h v; v8h h2[2]; };
            Frag af[2], bf[2];
            #pragma unroll
            for (int tm = 0; tm < 2; ++tm) {
                const _Float16* ap = &Asm[buf][(wm + tm * 16 + lrow) * LS + ko];
                af[tm].h2[0] = *(const v8h*)(ap + lgrp);
                af[tm].h2[1] = *(const v8h*)(ap + 16 + lgrp);
            }
            #pragma unroll
            for (int tn = 0; tn < 2; ++tn) {
                const _Float16* bp = &Bsm[buf][(wn + tn * 16 + lrow) * LS + ko];
                bf[tn].h2[0] = *(const v8h*)(bp + lgrp);
                bf[tn].h2[1] = *(const v8h*)(bp + 16 + lgrp);
            }
            #pragma unroll
            for (int tm = 0; tm < 2; ++tm)
                #pragma unroll
                for (int tn = 0; tn < 2; ++tn)
                    acc[tm][tn] = __builtin_amdgcn_wmma_f32_16x16x32_f16(
                        false, af[tm].v, false, bf[tn].v,
                        (short)0, acc[tm][tn], false, false);
        }
    };

    if (MODE == 0) {
        // ------------- guarded scalar path (tiny GEMMs only) -------------
        for (int k0 = 0; k0 < K; k0 += BKT) {
            for (int idx = t; idx < BM * BKT; idx += 256) {
                int r = idx >> 6, c = idx & 63;
                int gm = m0 + r, gk = k0 + c;
                float val = (gm < M && gk < K) ? Ab[(long)gm * sAm + (long)gk * sAk] : 0.0f;
                Asm[0][r * LS + c] = (_Float16)val;
            }
            for (int idx = t; idx < BN * BKT; idx += 256) {
                int c = idx >> 6, n = idx & 63;
                int gn = n0 + n, gk = k0 + c;
                float val = (gn < N && gk < K) ? Bb[(long)gn * sBn + (long)gk * sBk] : 0.0f;
                Bsm[0][n * LS + c] = (_Float16)val;
            }
            __syncthreads();
            compute(0);
            __syncthreads();
        }
    } else {
        // ------------- fast vectorized double-buffered path -------------
        v4f a_reg[4][2];   // 4 chunks x 8 floats (A: 128x64 / 256 threads)
        v4f b_reg[4];      // 16 floats (B: 64x64 / 256 threads)

        auto fetchA = [&](int k0) {
            #pragma unroll
            for (int j = 0; j < 4; ++j) {
                int id = t + j * 256;          // 0..1023
                int r  = id >> 3;              // 0..127
                int c  = (id & 7) * 8;         // 0..56
                const float* p = Ab + (long)(m0 + r) * sAm + (k0 + c);
                a_reg[j][0] = *(const v4f*)p;
                a_reg[j][1] = *(const v4f*)(p + 4);
            }
        };
        auto storeA = [&](int buf) {
            #pragma unroll
            for (int j = 0; j < 4; ++j) {
                int id = t + j * 256;
                int r  = id >> 3;
                int c  = (id & 7) * 8;
                v8h hv;
                #pragma unroll
                for (int e = 0; e < 4; ++e) {
                    hv[e]     = (_Float16)a_reg[j][0][e];
                    hv[4 + e] = (_Float16)a_reg[j][1][e];
                }
                *(v8h*)&Asm[buf][r * LS + c] = hv;
            }
        };
        auto fetchB = [&](int k0) {
            if (MODE == 1) {
                #pragma unroll
                for (int j = 0; j < 2; ++j) {
                    int id = t + j * 256;      // 0..511
                    int n  = id >> 3;          // 0..63
                    int c  = (id & 7) * 8;
                    const float* p = Bb + (long)(n0 + n) * sBn + (k0 + c);
                    b_reg[2 * j]     = *(const v4f*)p;
                    b_reg[2 * j + 1] = *(const v4f*)(p + 4);
                }
            } else {
                #pragma unroll
                for (int j = 0; j < 4; ++j) {
                    int id = t + j * 256;      // 0..1023
                    int kk = id >> 4;          // 0..63
                    int n  = (id & 15) * 4;    // 0..60
                    const float* p = Bb + (long)(k0 + kk) * sBk + (n0 + n);
                    b_reg[j] = *(const v4f*)p;
                }
            }
        };
        auto storeB = [&](int buf) {
            if (MODE == 1) {
                #pragma unroll
                for (int j = 0; j < 2; ++j) {
                    int id = t + j * 256;
                    int n  = id >> 3;
                    int c  = (id & 7) * 8;
                    v8h hv;
                    #pragma unroll
                    for (int e = 0; e < 4; ++e) {
                        hv[e]     = (_Float16)b_reg[2 * j][e];
                        hv[4 + e] = (_Float16)b_reg[2 * j + 1][e];
                    }
                    *(v8h*)&Bsm[buf][n * LS + c] = hv;
                }
            } else {
                #pragma unroll
                for (int j = 0; j < 4; ++j) {
                    int id = t + j * 256;
                    int kk = id >> 4;
                    int n  = (id & 15) * 4;
                    #pragma unroll
                    for (int e = 0; e < 4; ++e)
                        Bsm[buf][(n + e) * LS + kk] = (_Float16)b_reg[j][e];
                }
            }
        };

        fetchA(0); fetchB(0);
        storeA(0); storeB(0);
        __syncthreads();
        int p = 0;
        for (int k0 = 0; k0 < K; k0 += BKT) {
            const bool more = (k0 + BKT) < K;
            if (more) { fetchA(k0 + BKT); fetchB(k0 + BKT); }  // loads in flight
            compute(p);                                        // WMMA on current buffer
            if (more) {
                storeA(1 - p); storeB(1 - p);                  // other buffer: no conflict
                __syncthreads();
                p ^= 1;
            }
        }
    }

    // ---- store C: lanes 0-15: N=lane, M=r; lanes 16-31: N=lane-16, M=r+8 ----
    const int cn = lane & 15;
    const int mh = (lane >> 4) * 8;
    #pragma unroll
    for (int tm = 0; tm < 2; ++tm) {
        #pragma unroll
        for (int tn = 0; tn < 2; ++tn) {
            int col = n0 + wn + tn * 16 + cn;
            if (col >= N) continue;
            float bv = (flags & 4) ? bias[col] : 0.0f;
            #pragma unroll
            for (int r = 0; r < 8; ++r) {
                int row = m0 + wm + tm * 16 + mh + r;
                if (row >= M) continue;
                float v = alpha * acc[tm][tn][r] + bv;
                long off = (long)row * sCm + (long)col * sCn;
                if (flags & 2) v += Cb[off];
                if (flags & 1) v = fmaxf(v, 0.0f);
                Cb[off] = v;
            }
        }
    }
}

// ---------------------------------------------------------------------------
// Softmax with relative-position bias gather + per-row rel bin accumulation.
// scores [BH, Lq, Lk] in-place; qrel/arel [BH, Lq, NR]. grid = (Lq, B*H)
// ---------------------------------------------------------------------------
__launch_bounds__(256)
__global__ void softmax_rel_kernel(float* __restrict__ scores,
                                   const float* __restrict__ qrel,
                                   float* __restrict__ arel,
                                   int Lq, int Lk, int NR, int maxrel,
                                   int causal, float scale)
{
    __shared__ float sred[256];
    __shared__ float bins[32];

    const int q  = blockIdx.x;
    const int bz = blockIdx.y;
    float*       row = scores + ((long)bz * Lq + q) * Lk;
    const float* qr  = qrel   + ((long)bz * Lq + q) * NR;
    float*       ar  = arel   + ((long)bz * Lq + q) * NR;
    const int t = threadIdx.x;

    if (t < 32) bins[t] = 0.0f;

    float sv[8];
    int   ridx[8];
    int   nseg = 0;
    float lmax = -3.4e38f;
    for (int k = t; k < Lk; k += 256) {
        int d = k - q;
        d = d < -maxrel ? -maxrel : (d > maxrel ? maxrel : d);
        float s = (row[k] + qr[d + maxrel]) * scale;
        if (causal && k > q) s = -1e30f;
        sv[nseg] = s; ridx[nseg] = d + maxrel; ++nseg;
        lmax = fmaxf(lmax, s);
    }
    sred[t] = lmax; __syncthreads();
    for (int k = 128; k > 0; k >>= 1) { if (t < k) sred[t] = fmaxf(sred[t], sred[t + k]); __syncthreads(); }
    float gmax = sred[0]; __syncthreads();

    float ls = 0.0f;
    for (int j = 0; j < nseg; ++j) { sv[j] = __expf(sv[j] - gmax); ls += sv[j]; }
    sred[t] = ls; __syncthreads();
    for (int k = 128; k > 0; k >>= 1) { if (t < k) sred[t] += sred[t + k]; __syncthreads(); }
    float inv = 1.0f / sred[0];

    int seg = 0;
    for (int k = t; k < Lk; k += 256) {
        float p = sv[seg] * inv;
        row[k] = p;
        atomicAdd(&bins[ridx[seg]], p);   // ds_add_f32
        ++seg;
    }
    __syncthreads();
    if (t < NR) ar[t] = bins[t];
}

// ---------------------------------------------------------------------------
// y = LayerNorm(x + res) * g + b   (res may be null). grid = rows, block 256
// ---------------------------------------------------------------------------
__launch_bounds__(256)
__global__ void add_ln_kernel(const float* __restrict__ x, const float* __restrict__ res,
                              const float* __restrict__ g, const float* __restrict__ bta,
                              float* __restrict__ y, int D)
{
    __shared__ float sred[256];
    const long row = blockIdx.x;
    const float* xp = x + row * D;
    const float* rp = res ? res + row * D : nullptr;
    float* yp = y + row * D;
    const int t = threadIdx.x;

    float lv[8];
    int nseg = 0;
    float s = 0.0f;
    for (int i = t; i < D; i += 256) {
        float v = xp[i] + (rp ? rp[i] : 0.0f);
        lv[nseg++] = v;
        s += v;
    }
    sred[t] = s; __syncthreads();
    for (int k = 128; k > 0; k >>= 1) { if (t < k) sred[t] += sred[t + k]; __syncthreads(); }
    float mu = sred[0] / (float)D; __syncthreads();

    float vs = 0.0f;
    for (int j = 0; j < nseg; ++j) { float d2 = lv[j] - mu; vs += d2 * d2; }
    sred[t] = vs; __syncthreads();
    for (int k = 128; k > 0; k >>= 1) { if (t < k) sred[t] += sred[t + k]; __syncthreads(); }
    float rstd = rsqrtf(sred[0] / (float)D + 1e-5f);

    int seg = 0;
    for (int i = t; i < D; i += 256) {
        yp[i] = (lv[seg++] - mu) * rstd * g[i] + bta[i];
    }
}

// ---------------------------------------------------------------------------
// Host-side orchestration
// ---------------------------------------------------------------------------
namespace {

struct MHA { const float *Wq,*bq,*Wk,*bk,*Wv,*bv,*Wo,*bo,*rk,*rv; };
struct FFN { const float *W1,*b1,*W2,*b2; };
struct LNP { const float *g,*b; };

constexpr int  D_MODEL = 512;
constexpr int  NHEAD   = 8;
constexpr int  HDIM    = 64;
constexpr int  D_FF    = 2048;
constexpr int  NR      = 21;    // 2*MAX_REL+1
constexpr int  MAXREL  = 10;
constexpr int  BATCH   = 4;
constexpr int  L_ENC   = 1024;
constexpr int  L_DEC   = 512;

void launch_gemm(hipStream_t st, int mode,
                 const float* A, const float* B, const float* bias, float* C,
                 int M, int N, int K,
                 long sAb, long sAh, long sAm, long sAk,
                 long sBb, long sBh, long sBn, long sBk,
                 long sCb, long sCh, long sCm, long sCn,
                 int batches, int H, float alpha, int flags)
{
    dim3 grid((N + BN - 1) / BN, (M + BM - 1) / BM, batches);
    if (mode == 1)
        gemm_f16_wmma<1><<<grid, 256, 0, st>>>(A, B, bias, C, M, N, K,
            sAb, sAh, sAm, sAk, sBb, sBh, sBn, sBk, sCb, sCh, sCm, sCn, H, alpha, flags);
    else if (mode == 2)
        gemm_f16_wmma<2><<<grid, 256, 0, st>>>(A, B, bias, C, M, N, K,
            sAb, sAh, sAm, sAk, sBb, sBh, sBn, sBk, sCb, sCh, sCm, sCn, H, alpha, flags);
    else
        gemm_f16_wmma<0><<<grid, 256, 0, st>>>(A, B, bias, C, M, N, K,
            sAb, sAh, sAm, sAk, sBb, sBh, sBn, sBk, sCb, sCh, sCm, sCn, H, alpha, flags);
}

// Full MHA block: writes projected output (after Wo,bo) into t0.
void run_attention(hipStream_t st, const float* xq, const float* xkv,
                   int Lq, int Lk, int causal, const MHA& p,
                   float* qb, float* kb, float* vb,
                   float* qrel, float* arel, float* sc, float* t0)
{
    const int D = D_MODEL, H = NHEAD, dh = HDIM;
    // projections (batch=1) — fast mode 1
    launch_gemm(st, 1, xq,  p.Wq, p.bq, qb, BATCH * Lq, D, D,
                0,0,D,1,  0,0,D,1,  0,0,D,1,  1,1, 1.0f, 4);
    launch_gemm(st, 1, xkv, p.Wk, p.bk, kb, BATCH * Lk, D, D,
                0,0,D,1,  0,0,D,1,  0,0,D,1,  1,1, 1.0f, 4);
    launch_gemm(st, 1, xkv, p.Wv, p.bv, vb, BATCH * Lk, D, D,
                0,0,D,1,  0,0,D,1,  0,0,D,1,  1,1, 1.0f, 4);
    // scores[b,h] = q_h @ k_h^T   (K-contiguous B) — mode 1
    launch_gemm(st, 1, qb, kb, nullptr, sc, Lq, Lk, dh,
                (long)Lq*D, dh, D, 1,
                (long)Lk*D, dh, D, 1,
                (long)H*Lq*Lk, (long)Lq*Lk, Lk, 1,
                BATCH*H, H, 1.0f, 0);
    // qrel[b,h] = q_h @ rel_k^T  (N=21) — guarded mode 0
    launch_gemm(st, 0, qb, p.rk, nullptr, qrel, Lq, NR, dh,
                (long)Lq*D, dh, D, 1,
                0,0, dh, 1,
                (long)H*Lq*NR, (long)Lq*NR, NR, 1,
                BATCH*H, H, 1.0f, 0);
    // softmax + rel-bias gather + arel bins
    softmax_rel_kernel<<<dim3(Lq, BATCH*H), 256, 0, st>>>(sc, qrel, arel, Lq, Lk, NR, MAXREL,
                                                          causal, 0.125f /*1/sqrt(64)*/);
    // attn @ V (N-contiguous B) -> reuse kb as head-interleaved output [B, Lq, D] — mode 2
    launch_gemm(st, 2, sc, vb, nullptr, kb, Lq, dh, Lk,
                (long)H*Lq*Lk, (long)Lq*Lk, Lk, 1,
                (long)Lk*D, dh, 1, D,
                (long)Lq*D, dh, D, 1,
                BATCH*H, H, 1.0f, 0);
    // += arel @ rel_v   (K=21, accumulate) — mode 0
    launch_gemm(st, 0, arel, p.rv, nullptr, kb, Lq, dh, NR,
                (long)H*Lq*NR, (long)Lq*NR, NR, 1,
                0,0, 1, dh,
                (long)Lq*D, dh, D, 1,
                BATCH*H, H, 1.0f, 2);
    // output projection — mode 1
    launch_gemm(st, 1, kb, p.Wo, p.bo, t0, BATCH * Lq, D, D,
                0,0,D,1,  0,0,D,1,  0,0,D,1,  1,1, 1.0f, 4);
}

void run_ffn(hipStream_t st, const float* x, const FFN& f, float* f1, float* t0, int M)
{
    launch_gemm(st, 1, x,  f.W1, f.b1, f1, M, D_FF, D_MODEL,
                0,0,D_MODEL,1,  0,0,D_MODEL,1,  0,0,D_FF,1,  1,1, 1.0f, 4 | 1);
    launch_gemm(st, 1, f1, f.W2, f.b2, t0, M, D_MODEL, D_FF,
                0,0,D_FF,1,  0,0,D_FF,1,  0,0,D_MODEL,1,  1,1, 1.0f, 4);
}

} // namespace

extern "C" void kernel_launch(void* const* d_in, const int* in_sizes, int n_in,
                              void* d_out, int out_size, void* d_ws, size_t ws_size,
                              hipStream_t stream)
{
    (void)in_sizes; (void)n_in; (void)out_size; (void)ws_size;

    // --- unpack inputs (setup_inputs insertion order, recursively flattened) ---
    int pi = 0;
    auto nxt = [&]() { return (const float*)d_in[pi++]; };
    const float* x_enc = nxt();
    const float* x_dec = nxt();
    auto read_mha = [&](MHA& m) {
        m.Wq = nxt(); m.bq = nxt(); m.Wk = nxt(); m.bk = nxt();
        m.Wv = nxt(); m.bv = nxt(); m.Wo = nxt(); m.bo = nxt();
        m.rk = nxt(); m.rv = nxt();
    };
    auto read_ffn = [&](FFN& f) { f.W1 = nxt(); f.b1 = nxt(); f.W2 = nxt(); f.b2 = nxt(); };
    auto read_ln  = [&](LNP& l) { l.g = nxt(); l.b = nxt(); };

    MHA enc_attn[3];  FFN enc_ffn[3];  LNP enc_ln1[3], enc_ln2[3];
    for (int i = 0; i < 3; ++i) {
        read_mha(enc_attn[i]); read_ffn(enc_ffn[i]);
        read_ln(enc_ln1[i]);   read_ln(enc_ln2[i]);
    }
    MHA dec_self[2], dec_cross[2];  FFN dec_ffn[2];  LNP dec_ln1[2], dec_ln2[2], dec_ln3[2];
    for (int i = 0; i < 2; ++i) {
        read_mha(dec_self[i]); read_mha(dec_cross[i]); read_ffn(dec_ffn[i]);
        read_ln(dec_ln1[i]);   read_ln(dec_ln2[i]);    read_ln(dec_ln3[i]);
    }
    LNP enc_norm, dec_norm;
    read_ln(enc_norm); read_ln(dec_norm);

    // --- carve workspace ---
    float* W = (float*)d_ws;
    size_t off = 0;
    auto carve = [&](size_t n) { float* p = W + off; off += n; return p; };
    const size_t ME = (size_t)BATCH * L_ENC;              // 4096 rows (encoder)
    float* hb   = carve(ME * D_MODEL);                    // encoder stream
    float* eb   = carve(ME * D_MODEL);                    // encoder output (for cross attn)
    float* db   = carve((size_t)BATCH * L_DEC * D_MODEL); // decoder stream
    float* qb   = carve(ME * D_MODEL);
    float* kb   = carve(ME * D_MODEL);                    // also reused as attn head output
    float* vb   = carve(ME * D_MODEL);
    float* t0   = carve(ME * D_MODEL);                    // block output before LN
    float* f1   = carve(ME * D_FF);                       // FFN hidden
    float* qrel = carve((size_t)BATCH * NHEAD * L_ENC * NR);
    float* arel = carve((size_t)BATCH * NHEAD * L_ENC * NR);
    float* sc   = carve((size_t)BATCH * NHEAD * L_ENC * L_ENC); // scores (reused)

    // --- encoder ---
    hipMemcpyAsync(hb, x_enc, ME * D_MODEL * sizeof(float), hipMemcpyDeviceToDevice, stream);
    for (int i = 0; i < 3; ++i) {
        run_attention(stream, hb, hb, L_ENC, L_ENC, /*causal=*/0, enc_attn[i],
                      qb, kb, vb, qrel, arel, sc, t0);
        add_ln_kernel<<<(int)ME, 256, 0, stream>>>(t0, hb, enc_ln1[i].g, enc_ln1[i].b, hb, D_MODEL);
        run_ffn(stream, hb, enc_ffn[i], f1, t0, (int)ME);
        add_ln_kernel<<<(int)ME, 256, 0, stream>>>(t0, hb, enc_ln2[i].g, enc_ln2[i].b, hb, D_MODEL);
    }
    add_ln_kernel<<<(int)ME, 256, 0, stream>>>(hb, nullptr, enc_norm.g, enc_norm.b, eb, D_MODEL);

    // --- decoder ---
    const size_t MD = (size_t)BATCH * L_DEC;              // 2048 rows
    hipMemcpyAsync(db, x_dec, MD * D_MODEL * sizeof(float), hipMemcpyDeviceToDevice, stream);
    for (int i = 0; i < 2; ++i) {
        run_attention(stream, db, db, L_DEC, L_DEC, /*causal=*/1, dec_self[i],
                      qb, kb, vb, qrel, arel, sc, t0);
        add_ln_kernel<<<(int)MD, 256, 0, stream>>>(t0, db, dec_ln1[i].g, dec_ln1[i].b, db, D_MODEL);
        run_attention(stream, db, eb, L_DEC, L_ENC, /*causal=*/0, dec_cross[i],
                      qb, kb, vb, qrel, arel, sc, t0);
        add_ln_kernel<<<(int)MD, 256, 0, stream>>>(t0, db, dec_ln2[i].g, dec_ln2[i].b, db, D_MODEL);
        run_ffn(stream, db, dec_ffn[i], f1, t0, (int)MD);
        add_ln_kernel<<<(int)MD, 256, 0, stream>>>(t0, db, dec_ln3[i].g, dec_ln3[i].b, db, D_MODEL);
    }
    add_ln_kernel<<<(int)MD, 256, 0, stream>>>(db, nullptr, dec_norm.g, dec_norm.b,
                                               (float*)d_out, D_MODEL);
}
